// SGC_33208687133419
// MI455X (gfx1250) — matrix operand order repaired and checked by reading, not compiled, not run on hardware
//
#include <hip/hip_runtime.h>
#include <hip/hip_bf16.h>

typedef _Float16 half_t;
typedef __attribute__((ext_vector_type(16))) _Float16 v16h;
typedef __attribute__((ext_vector_type(8)))  _Float16 v8h;
typedef __attribute__((ext_vector_type(8)))  float    v8f;

#define DFEAT 128
#define DOUT  64

// ---------------- zero fill (float4 stores) ----------------
__global__ void k_zero_f4(float4* __restrict__ p, long long n4) {
  long long i  = (long long)blockIdx.x * blockDim.x + threadIdx.x;
  long long st = (long long)gridDim.x * blockDim.x;
  float4 z = make_float4(0.f, 0.f, 0.f, 0.f);
  for (; i < n4; i += st) p[i] = z;
}

// ---------------- f32 -> f16 convert ----------------
__global__ void k_f2h(const float* __restrict__ in, half_t* __restrict__ out, int n) {
  int i = blockIdx.x * blockDim.x + threadIdx.x;
  if (i < n) out[i] = (half_t)in[i];
}

// ---------------- SpMM hop: one wave per edge, 4 floats per lane ----------------
// hout[dst] += w * hin[src]  (row of 128 f32). Working set (51.2MB) fits in L2,
// so gathers + f32 atomics stay on-chip.
__global__ void k_spmm(const float* __restrict__ hin, float* __restrict__ hout,
                       const long long* __restrict__ ei, const float* __restrict__ ew,
                       long long E) {
  int lane = threadIdx.x & 31;
  long long w  = ((long long)blockIdx.x * blockDim.x + threadIdx.x) >> 5;
  long long nw = ((long long)gridDim.x * blockDim.x) >> 5;
  for (long long e = w; e < E; e += nw) {
    long long s = ei[e];
    long long d = ei[E + e];
    float wt = ew[e];
    if (e + nw < E) {  // prefetch next edge's source row -> global_prefetch_b8
      long long s2 = ei[e + nw];
      __builtin_prefetch(hin + s2 * DFEAT + lane * 4, 0, 0);
    }
    float4 v = *(const float4*)(hin + s * DFEAT + lane * 4);
    float* o = hout + d * DFEAT + lane * 4;
    atomicAdd(o + 0, v.x * wt);
    atomicAdd(o + 1, v.y * wt);
    atomicAdd(o + 2, v.z * wt);
    atomicAdd(o + 3, v.w * wt);
  }
}

// ---------------- gather representative rows, convert to f16 ----------------
__global__ void k_gather_h16(const float* __restrict__ h, const long long* __restrict__ rep,
                             half_t* __restrict__ hc, int C) {
  int tid = blockIdx.x * blockDim.x + threadIdx.x;
  if (tid >= C * 32) return;
  int c = tid >> 5, q = tid & 31;
  long long r = rep[c];
  float4 v = *(const float4*)(h + r * DFEAT + q * 4);
  half_t* o = hc + (long long)c * DFEAT + q * 4;
  o[0] = (half_t)v.x; o[1] = (half_t)v.y; o[2] = (half_t)v.z; o[3] = (half_t)v.w;
}

// A-fragment loader, contiguous K order (matches B):
// lane<16 holds row (lane), K = k0..k0+15; lane>=16 holds K = k0+16..k0+31.
__device__ __forceinline__ v16h ld_a_frag(const half_t* __restrict__ row, int k0, int lane) {
  int kb = k0 + ((lane >> 4) << 4);
  v8h lo = *(const v8h*)(row + kb);
  v8h hi = *(const v8h*)(row + kb + 8);
  return __builtin_shufflevector(lo, hi, 0,1,2,3,4,5,6,7,8,9,10,11,12,13,14,15);
}

// ---------------- GEMM1: h1 = relu(hc @ W1^T + b1), f16 out ----------------
// One wave per 16x16 tile; 4x v_wmma_f32_16x16x32_f16 over K=128.
__global__ void k_gemm1(const half_t* __restrict__ A, const half_t* __restrict__ W,
                        const float* __restrict__ bias, half_t* __restrict__ out, int C) {
  int lane = threadIdx.x & 31;
  int t = blockIdx.x * (blockDim.x >> 5) + (threadIdx.x >> 5);
  int ntile = (C >> 4) * (DFEAT >> 4);
  if (t >= ntile) return;
  int m0 = (t >> 3) << 4;            // DFEAT/16 == 8 column tiles
  int n0 = (t & 7) << 4;
  const half_t* arow = A + (long long)(m0 + (lane & 15)) * DFEAT;
  int n = n0 + (lane & 15);
  // B[k][n] = W[n][k]: lane reads 16 contiguous halves of W row n at K offset.
  const half_t* wrow = W + (long long)n * DFEAT + ((lane >> 4) << 4);
  v8f c = {};
#pragma unroll
  for (int k0 = 0; k0 < DFEAT; k0 += 32) {
    v16h a = ld_a_frag(arow, k0, lane);
    v16h b = *(const v16h*)(wrow + k0);
    c = __builtin_amdgcn_wmma_f32_16x16x32_f16(false, a, false, b, (short)0, c, false, false);
  }
  float bv = bias[n];
  int mb = m0 + ((lane >> 4) << 3);  // C/D layout: vgpr v -> row m = mb + v
#pragma unroll
  for (int v = 0; v < 8; ++v) {
    float r = c[v] + bv;
    r = r > 0.f ? r : 0.f;
    out[(long long)(mb + v) * DFEAT + n] = (half_t)r;
  }
}

// ---------------- GEMM2: y = h1 @ W2^T + b2, f32 out ----------------
__global__ void k_gemm2(const half_t* __restrict__ A, const half_t* __restrict__ W,
                        const float* __restrict__ bias, float* __restrict__ out, int C) {
  int lane = threadIdx.x & 31;
  int t = blockIdx.x * (blockDim.x >> 5) + (threadIdx.x >> 5);
  int ntile = (C >> 4) * (DOUT >> 4);
  if (t >= ntile) return;
  int m0 = (t >> 2) << 4;            // DOUT/16 == 4 column tiles
  int n0 = (t & 3) << 4;
  const half_t* arow = A + (long long)(m0 + (lane & 15)) * DFEAT;
  int n = n0 + (lane & 15);
  const half_t* wrow = W + (long long)n * DFEAT + ((lane >> 4) << 4);
  v8f c = {};
#pragma unroll
  for (int k0 = 0; k0 < DFEAT; k0 += 32) {
    v16h a = ld_a_frag(arow, k0, lane);
    v16h b = *(const v16h*)(wrow + k0);
    c = __builtin_amdgcn_wmma_f32_16x16x32_f16(false, a, false, b, (short)0, c, false, false);
  }
  float bv = bias[n];
  int mb = m0 + ((lane >> 4) << 3);
#pragma unroll
  for (int v = 0; v < 8; ++v)
    out[(long long)(mb + v) * DOUT + n] = c[v] + bv;
}

// ---------------- log_softmax in place over [C,64]: one wave per row ----------------
__global__ void k_lsm(float* __restrict__ y, int C) {
  int lane = threadIdx.x & 31;
  int w  = (blockIdx.x * blockDim.x + threadIdx.x) >> 5;
  int nw = (gridDim.x * blockDim.x) >> 5;
  for (int c = w; c < C; c += nw) {
    float v0 = y[(long long)c * DOUT + lane];
    float v1 = y[(long long)c * DOUT + 32 + lane];
    float m = fmaxf(v0, v1);
    for (int o = 16; o > 0; o >>= 1) m = fmaxf(m, __shfl_xor(m, o, 32));
    float s = __expf(v0 - m) + __expf(v1 - m);
    for (int o = 16; o > 0; o >>= 1) s += __shfl_xor(s, o, 32);
    float lse = m + __logf(s);
    y[(long long)c * DOUT + lane]      = v0 - lse;
    y[(long long)c * DOUT + 32 + lane] = v1 - lse;
  }
}

// ---------------- scatter per-cluster rows to all nodes ----------------
__global__ void k_scatter(const float* __restrict__ y, const long long* __restrict__ ci,
                          float* __restrict__ out, int N) {
  long long tid = (long long)blockIdx.x * blockDim.x + threadIdx.x;
  if (tid >= (long long)N * 16) return;
  long long nn = tid >> 4;
  int q = (int)(tid & 15);
  long long c = ci[nn];
  float4 v = *(const float4*)(y + c * DOUT + q * 4);
  *(float4*)(out + nn * DOUT + q * 4) = v;
}

extern "C" void kernel_launch(void* const* d_in, const int* in_sizes, int n_in,
                              void* d_out, int out_size, void* d_ws, size_t ws_size,
                              hipStream_t stream) {
  const float*     x   = (const float*)d_in[0];
  const long long* ei  = (const long long*)d_in[1];   // [2,E] int64
  const float*     ew  = (const float*)d_in[2];
  const long long* rep = (const long long*)d_in[3];   // [C] int64
  const long long* ci  = (const long long*)d_in[4];   // [N] int64
  const float*     W1  = (const float*)d_in[5];
  const float*     b1  = (const float*)d_in[6];
  const float*     W2  = (const float*)d_in[7];
  const float*     b2  = (const float*)d_in[8];
  float* out = (float*)d_out;

  const long long N = in_sizes[0] / DFEAT;   // 100000
  const long long E = in_sizes[2];           // 1600000
  const int       C = in_sizes[3];           // 50000

  // workspace carve-out (256B aligned)
  char* ws = (char*)d_ws;
  size_t o = 0;
  auto take = [&](size_t bytes) -> char* {
    char* p = ws + o;
    o += (bytes + 255) & ~(size_t)255;
    return p;
  };
  float*  hA  = (float*) take((size_t)N * DFEAT * 4);
  float*  hB  = (float*) take((size_t)N * DFEAT * 4);
  half_t* hc  = (half_t*)take((size_t)C * DFEAT * 2);
  half_t* h1  = (half_t*)take((size_t)C * DFEAT * 2);
  float*  y   = (float*) take((size_t)C * DOUT  * 4);
  half_t* W1h = (half_t*)take((size_t)DFEAT * DFEAT * 2);
  half_t* W2h = (half_t*)take((size_t)DOUT  * DFEAT * 2);

  // weight conversion
  k_f2h<<<(DFEAT * DFEAT + 255) / 256, 256, 0, stream>>>(W1, W1h, DFEAT * DFEAT);
  k_f2h<<<(DOUT  * DFEAT + 255) / 256, 256, 0, stream>>>(W2, W2h, DOUT * DFEAT);

  // two SpMM hops (zero destination each time; ws is not re-poisoned but we re-zero per call)
  long long n4 = N * DFEAT / 4;
  k_zero_f4<<<2048, 256, 0, stream>>>((float4*)hA, n4);
  k_spmm<<<8192, 256, 0, stream>>>(x, hA, ei, ew, E);
  k_zero_f4<<<2048, 256, 0, stream>>>((float4*)hB, n4);
  k_spmm<<<8192, 256, 0, stream>>>(hA, hB, ei, ew, E);

  // gather compressed rows -> f16
  k_gather_h16<<<(C * 32 + 255) / 256, 256, 0, stream>>>(hB, rep, hc, C);

  // MLP via WMMA
  int tiles1 = (C / 16) * (DFEAT / 16);
  k_gemm1<<<(tiles1 + 3) / 4, 128, 0, stream>>>(hc, W1h, b1, h1, C);
  int tiles2 = (C / 16) * (DOUT / 16);
  k_gemm2<<<(tiles2 + 3) / 4, 128, 0, stream>>>(h1, W2h, b2, y, C);

  // log_softmax per cluster, then broadcast to nodes
  k_lsm<<<1024, 256, 0, stream>>>(y, C);
  k_scatter<<<(int)(((long long)N * 16 + 255) / 256), 256, 0, stream>>>(y, ci, out, (int)N);
}
